// ChannelWiseTCN_20607253086711
// MI455X (gfx1250) — matrix operand order repaired and checked by reading, not compiled, not run on hardware
//
#include <hip/hip_runtime.h>

// ---------------------------------------------------------------------------
// Fused 4-layer TCN for MI455X (gfx1250, wave32).
//  - Whole activation tile resident in LDS across all layers (halo recompute).
//  - Pointwise convs via v_wmma_f32_16x16x32_bf16 with split-bf16 (hi+lo)
//    operands for ~fp32 accuracy at bf16-WMMA rate.
//  - Layer loop fully unrolled: layer-0 dual-accumulator GEMM is branch-free.
//  - x tile staged with async Global->LDS (ASYNCcnt-tracked
//    global_load_async_to_lds_b128) -- builtin confirmed present.
//  - HBM traffic ~ read x once + write out once  =>  ~1.5us at 23.3 TB/s.
// ---------------------------------------------------------------------------

#define LT   64          // output seq positions per workgroup
#define HALO 32          // receptive-field halo (actual need: 30)
#define NP   (LT + HALO) // 96 tile positions computed per layer
#define HS   100         // hbuf stride (floats), padded vs 96 for banks
#define YS   136         // y-plane channel stride (halfwords), padded vs 128
#define NCH  128         // hidden channels H
#define SEQL 4096
#define NCHUNK (SEQL / LT) // 64

#if __has_builtin(__builtin_amdgcn_global_load_async_to_lds_b128) && \
    __has_builtin(__builtin_amdgcn_s_wait_asynccnt)
#define HAVE_ASYNC_LDS 1
#else
#define HAVE_ASYNC_LDS 0
#endif

typedef int v4i __attribute__((vector_size(16)));            // GCC-style vector, per builtin proto
typedef __attribute__((address_space(1))) v4i* gptr_v4i;     // global
typedef __attribute__((address_space(3))) v4i* lptr_v4i;     // LDS

typedef __attribute__((ext_vector_type(16))) __bf16 v16bf;
typedef __attribute__((ext_vector_type(8)))  __bf16 v8bf;
typedef __attribute__((ext_vector_type(8)))  float  v8f;

__global__ __launch_bounds__(256)
void tcn_fused_kernel(const float* __restrict__ x,        // (32,64,4096)
                      const float* __restrict__ sc_w,     // (4,3)
                      const float* __restrict__ pw_w0,    // (128,64)
                      const float* __restrict__ pw_rest,  // (3,128,128)
                      const float* __restrict__ pw_b,     // (4,128)
                      const float* __restrict__ bn_g,     // (4,128)
                      const float* __restrict__ bn_b,     // (4,128)
                      const float* __restrict__ bn_m,     // (4,128)
                      const float* __restrict__ bn_v,     // (4,128)
                      const float* __restrict__ res_w,    // (128,64)
                      const float* __restrict__ res_b,    // (128)
                      const float* __restrict__ out_w,    // (128)
                      const float* __restrict__ out_b,    // (1)
                      float* __restrict__ out)            // (32,1,4096)
{
    __shared__ __align__(64) float  hbuf[NCH * HS];   // resident activations (fp32)
    __shared__ __align__(64) __bf16 yhi[NP * YS];     // GEMM B operand, hi part [t][c]
    __shared__ __align__(64) __bf16 ylo[NP * YS];     // GEMM B operand, lo part [t][c]
    __shared__ __align__(64) float  red[4 * LT];      // output-projection partials

    const int blk   = blockIdx.x;
    const int b     = blk >> 6;          // batch
    const int chunk = blk & (NCHUNK - 1);
    const int t0    = chunk * LT;        // global output start
    const int tid   = threadIdx.x;
    const int lane  = tid & 31;
    const int wave  = tid >> 5;
    const int hh    = lane >> 4;         // lane half (0/1)
    const int ln16  = lane & 15;

    // ---- stage x halo tile into hbuf channels 0..63 (B128 granularity) -----
    #pragma unroll 1
    for (int idx = tid; idx < 64 * (NP / 4); idx += 256) {
        const int c = idx / (NP / 4);
        const int t = (idx - c * (NP / 4)) * 4;      // 4-aligned; halo is 4-aligned
        const int gt = t0 - HALO + t;
        if (gt >= 0) {
#if HAVE_ASYNC_LDS
            __builtin_amdgcn_global_load_async_to_lds_b128(
                (gptr_v4i)(uintptr_t)(x + (b * 64 + c) * SEQL + gt),
                (lptr_v4i)(uintptr_t)(&hbuf[c * HS + t]), 0, 0);
#else
            const float4 v = *(const float4*)(x + (b * 64 + c) * SEQL + gt);
            *(float4*)(&hbuf[c * HS + t]) = v;
#endif
        } else {
            *(float4*)(&hbuf[c * HS + t]) = make_float4(0.f, 0.f, 0.f, 0.f);
        }
    }
#if HAVE_ASYNC_LDS
    __builtin_amdgcn_s_wait_asynccnt(0);
#endif
    // hint the next chunk into L2 (emits global_prefetch_b8)
    if (chunk + 1 < NCHUNK && tid < 64)
        __builtin_prefetch(&x[(b * 64 + tid) * SEQL + t0 + LT], 0, 1);

    #pragma unroll
    for (int layer = 0; layer < 4; ++layer) {
        const int dil = 1 << layer;
        const float w0 = sc_w[layer * 3 + 0];
        const float w1 = sc_w[layer * 3 + 1];
        const float w2 = sc_w[layer * 3 + 2];
        __syncthreads();

        // ---- step 1: tied-kernel causal depthwise conv + bf16 hi/lo split --
        // layer 0: rows 0..63 = dwconv(x), rows 64..127 = raw x (residual GEMM)
        #pragma unroll 1
        for (int idx = tid; idx < NCH * NP; idx += 256) {
            const int c = idx / NP, t = idx - c * NP;
            float v;
            if (layer == 0 && c >= 64) {
                v = hbuf[(c - 64) * HS + t];
            } else {
                const float a0 = hbuf[c * HS + t];
                const float a1 = (t >= dil)     ? hbuf[c * HS + t - dil]     : 0.0f;
                const float a2 = (t >= 2 * dil) ? hbuf[c * HS + t - 2 * dil] : 0.0f;
                v = w2 * a0 + w1 * a1 + w0 * a2;
            }
            const __bf16 hi = (__bf16)v;
            const __bf16 lo = (__bf16)(v - (float)hi);
            yhi[t * YS + c] = hi;
            ylo[t * YS + c] = lo;
        }
        __syncthreads();

        // ---- step 2: WMMA GEMM (M=128, N=96, K=128) + fused epilogue -------
        const int mt = wave;               // each wave owns one 16-row M tile
        const int m  = mt * 16 + ln16;     // A-matrix row held by this lane

        // Build split-bf16 A fragments for all 4 K-steps.
        // 16-bit A layout (ISA 7.12.2): lane m holds K in [8h,8h+8) and [16+8h,16+8h+8).
        v16bf ahi[4], alo[4];
        #pragma unroll
        for (int ks = 0; ks < 4; ++ks) {
            const float* Wrow;
            int koff;
            if (layer == 0) {              // ks 0,1 -> pw_w0 ; ks 2,3 -> res_w
                Wrow = ((ks < 2) ? pw_w0 : res_w) + m * 64;
                koff = (ks & 1) * 32;
            } else {
                Wrow = pw_rest + (layer - 1) * NCH * NCH + m * NCH;
                koff = ks * 32;
            }
            const float* p0 = Wrow + koff + 8 * hh;
            const float* p1 = Wrow + koff + 16 + 8 * hh;
            v16bf ah, al;
            #pragma unroll
            for (int e = 0; e < 8; ++e) {
                float f = p0[e];
                __bf16 h = (__bf16)f;
                ah[e] = h; al[e] = (__bf16)(f - (float)h);
            }
            #pragma unroll
            for (int e = 0; e < 8; ++e) {
                float f = p1[e];
                __bf16 h = (__bf16)f;
                ah[8 + e] = h; al[8 + e] = (__bf16)(f - (float)h);
            }
            ahi[ks] = ah; alo[ks] = al;
        }

        // Fold BN+bias into scale/shift per output channel (per r slot).
        float sA[8], sB[8], rB[8];
        #pragma unroll
        for (int r = 0; r < 8; ++r) {
            const int co = mt * 16 + r + 8 * hh;
            const float g   = bn_g[layer * NCH + co];
            const float be  = bn_b[layer * NCH + co];
            const float mu  = bn_m[layer * NCH + co];
            const float var = bn_v[layer * NCH + co];
            const float inv = __frsqrt_rn(var + 1e-5f);
            sA[r] = g * inv;
            sB[r] = g * inv * (pw_b[layer * NCH + co] - mu) + be;
            rB[r] = (layer == 0) ? res_b[co] : 0.0f;
        }

        #pragma unroll 1
        for (int nt = 0; nt < NP / 16; ++nt) {     // 6 N tiles per wave
            const int t = nt * 16 + ln16;          // B/D column held by this lane
            v8f acc1 = {};                         // pointwise path
            v8f acc2 = {};                         // layer-0 residual 1x1 path
            #pragma unroll
            for (int ks = 0; ks < 4; ++ks) {
                // B layout: lane n holds K in [8h,8h+8) and [16+8h,16+8h+8), c-contiguous in LDS
                const __bf16* bph = &yhi[t * YS + ks * 32 + 8 * hh];
                const __bf16* bpl = &ylo[t * YS + ks * 32 + 8 * hh];
                v8bf b0 = *(const v8bf*)(bph);
                v8bf b1 = *(const v8bf*)(bph + 16);
                v8bf b2 = *(const v8bf*)(bpl);
                v8bf b3 = *(const v8bf*)(bpl + 16);
                v16bf bvh = __builtin_shufflevector(b0, b1, 0,1,2,3,4,5,6,7,8,9,10,11,12,13,14,15);
                v16bf bvl = __builtin_shufflevector(b2, b3, 0,1,2,3,4,5,6,7,8,9,10,11,12,13,14,15);
                if (layer == 0 && ks >= 2) {       // compile-time after unroll
                    acc2 = __builtin_amdgcn_wmma_f32_16x16x32_bf16(false, ahi[ks], false, bvh, (short)0, acc2, false, false);
                    acc2 = __builtin_amdgcn_wmma_f32_16x16x32_bf16(false, ahi[ks], false, bvl, (short)0, acc2, false, false);
                    acc2 = __builtin_amdgcn_wmma_f32_16x16x32_bf16(false, alo[ks], false, bvh, (short)0, acc2, false, false);
                } else {
                    acc1 = __builtin_amdgcn_wmma_f32_16x16x32_bf16(false, ahi[ks], false, bvh, (short)0, acc1, false, false);
                    acc1 = __builtin_amdgcn_wmma_f32_16x16x32_bf16(false, ahi[ks], false, bvl, (short)0, acc1, false, false);
                    acc1 = __builtin_amdgcn_wmma_f32_16x16x32_bf16(false, alo[ks], false, bvh, (short)0, acc1, false, false);
                }
            }
            // Epilogue: BN -> ReLU -> +residual -> ReLU, write back in place.
            // D layout: VGPR r holds (M = r + 8h, N = lane%16).
            #pragma unroll
            for (int r = 0; r < 8; ++r) {
                const int co = mt * 16 + r + 8 * hh;
                float z = sA[r] * acc1[r] + sB[r];
                z = fmaxf(z, 0.0f);
                const float rr = (layer == 0) ? (acc2[r] + rB[r])
                                              : hbuf[co * HS + t];
                hbuf[co * HS + t] = fmaxf(z + rr, 0.0f);
            }
        }
    }

    __syncthreads();
    // ---- output projection: (1,128) . h -> (B,1,L), all 256 threads --------
    {
        const int tpos = tid & (LT - 1);     // 0..63 output position
        const int part = tid >> 6;           // 0..3 channel partition
        const int t = HALO + tpos;
        float s = 0.0f;
        #pragma unroll 8
        for (int c = part * 32; c < part * 32 + 32; ++c)
            s += out_w[c] * hbuf[c * HS + t];
        red[part * LT + tpos] = s;
    }
    __syncthreads();
    if (tid < LT) {
        const float s = out_b[0] + red[tid] + red[LT + tid] +
                        red[2 * LT + tid] + red[3 * LT + tid];
        out[b * SEQL + t0 + tid] = s;
    }
}

extern "C" void kernel_launch(void* const* d_in, const int* in_sizes, int n_in,
                              void* d_out, int out_size, void* d_ws, size_t ws_size,
                              hipStream_t stream) {
    (void)in_sizes; (void)n_in; (void)d_ws; (void)ws_size; (void)out_size;
    const float* x       = (const float*)d_in[0];
    const float* sc_w    = (const float*)d_in[1];
    const float* pw_w0   = (const float*)d_in[2];
    const float* pw_rest = (const float*)d_in[3];
    const float* pw_b    = (const float*)d_in[4];
    const float* bn_g    = (const float*)d_in[5];
    const float* bn_b    = (const float*)d_in[6];
    const float* bn_m    = (const float*)d_in[7];
    const float* bn_v    = (const float*)d_in[8];
    const float* res_w   = (const float*)d_in[9];
    const float* res_b   = (const float*)d_in[10];
    const float* out_w   = (const float*)d_in[11];
    const float* out_b   = (const float*)d_in[12];

    dim3 grid(32 * NCHUNK);   // 2048 workgroups: (batch, seq-chunk)
    dim3 block(256);          // 8 wave32 waves; ~104 KB LDS -> up to 3 WGs/WGP
    hipLaunchKernelGGL(tcn_fused_kernel, grid, block, 0, stream,
                       x, sc_w, pw_w0, pw_rest, pw_b, bn_g, bn_b, bn_m, bn_v,
                       res_w, res_b, out_w, out_b, (float*)d_out);
}
